// CrossAttentionSubLayer_24764781428984
// MI455X (gfx1250) — matrix operand order repaired and checked by reading, not compiled, not run on hardware
//
#include <hip/hip_runtime.h>
#include <hip/hip_bf16.h>

#define MODEL_DIM 1024
#define EMBED_DIM 1024
#define NB 4
#define SQN 2048
#define SKN 2048
#define S_STRIDE 2052   // f32 scores row stride (pad vs banks)
#define P_STRIDE 2056   // bf16 probs row stride (multiple of 8 -> 16B-aligned rows)

typedef __attribute__((ext_vector_type(16))) __bf16 v16bf;
typedef __attribute__((ext_vector_type(8)))  float  v8f;

#define WMMA_BF16(a, b, c) \
  __builtin_amdgcn_wmma_f32_16x16x32_bf16(false, (a), false, (b), (short)0, (c), false, false)

// ---- f32 -> packed bf16x2 (native v_cvt_pk_bf16_f32 when the builtin exists) ----
#if __has_builtin(__builtin_amdgcn_cvt_pk_bf16_f32)
typedef __attribute__((ext_vector_type(2))) __bf16 v2bf;
__device__ __forceinline__ unsigned pk_bf16(float a, float b) {
  union { v2bf v; unsigned u; } c;
  c.v = __builtin_amdgcn_cvt_pk_bf16_f32(a, b);
  return c.u;
}
#else
__device__ __forceinline__ unsigned pk_bf16(float a, float b) {
  unsigned ua = __float_as_uint(a);
  unsigned ub = __float_as_uint(b);
  ua = (ua + 0x7FFFu + ((ua >> 16) & 1u)) >> 16;   // RNE
  ub = (ub + 0x7FFFu + ((ub >> 16) & 1u)) >> 16;
  return (ua & 0xFFFFu) | (ub << 16);
}
#endif

__device__ __forceinline__ unsigned short f2bf(float a) {
  return (unsigned short)(pk_bf16(a, 0.0f) & 0xFFFFu);
}

// A-matrix 16x32 bf16 fragment (ISA 7.12.2): lane half h selects K offset 8.
// Elements are two contiguous 16B runs: p[8h .. 8h+7] and p[16+8h .. 23+8h].
__device__ __forceinline__ v16bf fragA_bf16(const unsigned short* p, int h) {
  union { v16bf v; uint4 q[2]; } r;
  r.q[0] = *(const uint4*)(p + 8 * h);
  r.q[1] = *(const uint4*)(p + 16 + 8 * h);
  return r.v;
}

// B-matrix 32x16 bf16 fragment: lane half h selects K offset 8.
// Elements are two contiguous 16B runs: p[8h .. 8h+7] and p[16+8h .. 23+8h].
__device__ __forceinline__ v16bf fragB_bf16(const unsigned short* p, int h) {
  union { v16bf v; uint4 q[2]; } r;
  r.q[0] = *(const uint4*)(p + 8 * h);
  r.q[1] = *(const uint4*)(p + 16 + 8 * h);
  return r.v;
}

// ---------------------------------------------------------------------------
// Prep: convert f32 tensor -> bf16 (optionally scaled). Memory-bound sweep.
// ---------------------------------------------------------------------------
__global__ __launch_bounds__(256) void cvt_kernel(const float* __restrict__ src,
                                                  unsigned* __restrict__ dst,
                                                  int npairs, float scale) {
  int i = blockIdx.x * 256 + threadIdx.x;
  if (i < npairs) {
    float2 t = ((const float2*)src)[i];
    dst[i] = pk_bf16(t.x * scale, t.y * scale);
  }
}

// Transpose + convert W [D,E] f32 -> Wt [E,D] bf16 (B-fragment friendly)
__global__ __launch_bounds__(256) void wt_kernel(const float* __restrict__ W,
                                                 unsigned short* __restrict__ Wt) {
  int idx = blockIdx.x * 256 + threadIdx.x;      // over 1024*1024
  int n = idx >> 10;
  int k = idx & 1023;
  Wt[(size_t)n * MODEL_DIM + k] = f2bf(W[(size_t)k * EMBED_DIM + n]);
}

// ---------------------------------------------------------------------------
// Fused attention. One workgroup = (batch, 16 q-rows), 8 waves.
// Pass A: scores -> LDS f32. Softmax -> LDS bf16 P. Pass B: P @ V^T -> ws bf16.
// ---------------------------------------------------------------------------
__global__ __launch_bounds__(256) void attn_kernel(
    const unsigned short* __restrict__ Qbf, const unsigned short* __restrict__ Kbf,
    const unsigned short* __restrict__ Vbf, const int* __restrict__ kmask,
    unsigned short* __restrict__ resb) {
  extern __shared__ unsigned char smem[];
  unsigned short* P = (unsigned short*)smem;                     // [16][P_STRIDE] bf16
  float* S = (float*)(smem + 16 * P_STRIDE * 2);                 // [16][S_STRIDE] f32
  float* rowinv = (float*)(smem + 16 * P_STRIDE * 2 + 16 * S_STRIDE * 4);  // [16]

  const int wave = threadIdx.x >> 5;
  const int lane = threadIdx.x & 31;
  const int h = lane >> 4;
  const int lr = lane & 15;

  const int b  = blockIdx.x >> 7;     // SQN/16 = 128 tiles per batch
  const int q0 = (blockIdx.x & 127) * 16;

  const v8f vz = {0.f, 0.f, 0.f, 0.f, 0.f, 0.f, 0.f, 0.f};

  // ---------------- Pass A: S = Qs K^T (1/sqrt(d) pre-folded into Qbf) ------
  const unsigned short* Qrow = Qbf + (size_t)(b * SQN + q0 + lr) * MODEL_DIM;
  const int kcol0 = wave * 256;                  // this wave's 256 key columns

  v8f acc[16];
#pragma unroll
  for (int t = 0; t < 16; ++t) acc[t] = vz;

  for (int d0 = 0; d0 < MODEL_DIM; d0 += 32) {
    v16bf a = fragA_bf16(Qrow + d0, h);          // 2x b128, reused for 16 WMMAs
#pragma unroll
    for (int t = 0; t < 16; ++t) {
      const unsigned short* Kp =
          Kbf + (size_t)(b * SKN + kcol0 + t * 16 + lr) * MODEL_DIM + d0;
      v16bf bb = fragB_bf16(Kp, h);              // 2x b128, contiguous in d
      acc[t] = WMMA_BF16(a, bb, acc[t]);
    }
  }

  // store scores with reference mask semantics (+1e-13, masked -> -1e-13)
#pragma unroll
  for (int t = 0; t < 16; ++t) {
    int col = kcol0 + t * 16 + lr;
    int mv = kmask[b * SKN + col];
#pragma unroll
    for (int r = 0; r < 8; ++r) {
      float v = acc[t][r] + 1e-13f;
      if (mv == 0) v = -1e-13f;
      S[(r + 8 * h) * S_STRIDE + col] = v;
    }
  }
  __syncthreads();

  // ---------------- softmax: wave per 2 rows; emit bf16 P ----------------
#pragma unroll
  for (int rr = 0; rr < 2; ++rr) {
    int row = wave * 2 + rr;
    float* Sr = S + row * S_STRIDE;
    unsigned short* Pr = P + row * P_STRIDE;
    float m = -3.0e38f;
    for (int c = lane * 2; c < SKN; c += 64) {
      float2 t = *(const float2*)(Sr + c);
      m = fmaxf(m, fmaxf(t.x, t.y));
    }
#pragma unroll
    for (int off = 16; off > 0; off >>= 1) m = fmaxf(m, __shfl_xor(m, off, 32));
    float s = 0.f;
    for (int c = lane * 2; c < SKN; c += 64) {
      float2 t = *(const float2*)(Sr + c);
      float e0 = __expf(t.x - m);
      float e1 = __expf(t.y - m);
      s += e0 + e1;
      *(unsigned*)(Pr + c) = pk_bf16(e0, e1);
    }
#pragma unroll
    for (int off = 16; off > 0; off >>= 1) s += __shfl_xor(s, off, 32);
    if (lane == 0) rowinv[row] = 1.0f / s;
  }
  __syncthreads();

  // ---------------- Pass B: res = P V^T  (Vbf is [B, D, SK]) ----------------
  const unsigned short* Vb = Vbf + (size_t)b * MODEL_DIM * SKN;
  const int c0 = wave * 128;                     // this wave's 128 d-columns
  v8f acc2[8];
#pragma unroll
  for (int n = 0; n < 8; ++n) acc2[n] = vz;

  for (int k0 = 0; k0 < SKN; k0 += 32) {
    v16bf a = fragA_bf16(P + lr * P_STRIDE + k0, h);   // 2x ds_load_b128
#pragma unroll
    for (int n = 0; n < 8; ++n) {
      const unsigned short* Vp = Vb + (size_t)(c0 + n * 16 + lr) * SKN + k0;
      v16bf bb = fragB_bf16(Vp, h);                    // contiguous in k
      acc2[n] = WMMA_BF16(a, bb, acc2[n]);
    }
  }

  // scale by 1/rowsum and store bf16 result to workspace
#pragma unroll
  for (int r = 0; r < 8; ++r) {
    int m = r + 8 * h;
    float rinv = rowinv[m];
    size_t base = (size_t)(b * SQN + q0 + m) * MODEL_DIM;
#pragma unroll
    for (int n = 0; n < 8; ++n) {
      resb[base + c0 + n * 16 + lr] = f2bf(acc2[n][r] * rinv);
    }
  }
}

// ---------------------------------------------------------------------------
// out = res @ W + b. 128x64 tile per workgroup, wave per 16-row strip.
// ---------------------------------------------------------------------------
__global__ __launch_bounds__(256) void proj_kernel(
    const unsigned short* __restrict__ resb, const unsigned short* __restrict__ Wt,
    const float* __restrict__ bias, float* __restrict__ out) {
  const int wave = threadIdx.x >> 5;
  const int lane = threadIdx.x & 31;
  const int h = lane >> 4;
  const int lr = lane & 15;
  const int r0 = blockIdx.x * 128 + wave * 16;
  const int c0 = blockIdx.y * 64;

  const v8f vz = {0.f, 0.f, 0.f, 0.f, 0.f, 0.f, 0.f, 0.f};
  v8f acc[4];
#pragma unroll
  for (int n = 0; n < 4; ++n) acc[n] = vz;

  for (int k0 = 0; k0 < MODEL_DIM; k0 += 32) {
    v16bf a = fragA_bf16(resb + (size_t)(r0 + lr) * MODEL_DIM + k0, h);
#pragma unroll
    for (int n = 0; n < 4; ++n) {
      v16bf bb = fragB_bf16(Wt + (size_t)(c0 + n * 16 + lr) * MODEL_DIM + k0, h);
      acc[n] = WMMA_BF16(a, bb, acc[n]);
    }
  }

#pragma unroll
  for (int n = 0; n < 4; ++n) {
    int cc = c0 + n * 16 + lr;
    float bv = bias[cc];
#pragma unroll
    for (int r = 0; r < 8; ++r) {
      out[(size_t)(r0 + r + 8 * h) * EMBED_DIM + cc] = acc[n][r] + bv;
    }
  }
}

extern "C" void kernel_launch(void* const* d_in, const int* in_sizes, int n_in,
                              void* d_out, int out_size, void* d_ws, size_t ws_size,
                              hipStream_t stream) {
  const float* Q    = (const float*)d_in[0];
  const float* K    = (const float*)d_in[1];
  const float* V    = (const float*)d_in[2];
  // d_in[3] = query_attention_mask (unused by reference math)
  const int*   kmsk = (const int*)d_in[4];
  const float* W    = (const float*)d_in[5];
  const float* bias = (const float*)d_in[6];
  float* out = (float*)d_out;

  const size_t TENS = (size_t)NB * SQN * MODEL_DIM;   // 8M elements
  // ws layout (bf16): Qbf[0,16M) Kbf[16M,32M) Vbf[32M,48M) res[48M,64M) Wt[64M,66M)
  unsigned short* Qbf  = (unsigned short*)d_ws;
  unsigned short* Kbf  = Qbf + TENS;
  unsigned short* Vbf  = Kbf + TENS;
  unsigned short* resb = Vbf + TENS;
  unsigned short* Wt   = resb + TENS;

  const int npairs = (int)(TENS / 2);                 // 4M float2 per tensor
  const int cvtBlocks = npairs / 256;
  cvt_kernel<<<dim3(cvtBlocks), dim3(256), 0, stream>>>(Q, (unsigned*)Qbf, npairs,
                                                        0.03125f);  // 1/sqrt(1024)
  cvt_kernel<<<dim3(cvtBlocks), dim3(256), 0, stream>>>(K, (unsigned*)Kbf, npairs, 1.0f);
  cvt_kernel<<<dim3(cvtBlocks), dim3(256), 0, stream>>>(V, (unsigned*)Vbf, npairs, 1.0f);
  wt_kernel<<<dim3((MODEL_DIM * EMBED_DIM) / 256), dim3(256), 0, stream>>>(W, Wt);

  size_t shmem = (size_t)16 * P_STRIDE * 2 + (size_t)16 * S_STRIDE * 4 + 64;
  attn_kernel<<<dim3(NB * (SQN / 16)), dim3(256), shmem, stream>>>(Qbf, Kbf, Vbf, kmsk,
                                                                   resb);

  proj_kernel<<<dim3((NB * SQN) / 128, EMBED_DIM / 64), dim3(256), 0, stream>>>(
      resb, Wt, bias, out);
}
